// HGraphConvolution_36807869727112
// MI455X (gfx1250) — matrix operand order repaired and checked by reading, not compiled, not run on hardware
//
#include <hip/hip_runtime.h>
#include <hip/hip_bf16.h>

// HGCN HypLinear + HypAgg + HypAct for MI455X (gfx1250, wave32, WMMA).
// Pipeline (all on `stream`, deterministic, no atomics):
//   K0 hgcn_bias   : hyp_bias = proj(expmap0(bias)) -> ws
//   Ks hgcn_split  : x,W -> (hi,lo) fp16 pairs in ws (split hoisted out of GEMM)
//   K1 hgcn_gemm   : mx = x @ W^T via v_wmma_f32_16x16x32_f16, 3 WMMA per
//                    32-K chunk (hi*hi + hi*lo + lo*hi => ~fp32 accuracy),
//                    inner loop = 8x global_load_b128 + 3x wmma
//   K2 hgcn_hyplinear_post : mobius scaling + proj + mobius_add + proj + logmap0
//   K3 hgcn_agg    : deterministic segment-sum (sorted adj_row, binary search)
//   K4 hgcn_act    : expmap0/proj/logmap0/relu/expmap0/proj (in-place on out)

typedef __attribute__((ext_vector_type(16))) _Float16 v16h;
typedef __attribute__((ext_vector_type(8)))  _Float16 v8h;
typedef __attribute__((ext_vector_type(8)))  float    v8f;

#define N_NODES   10000
#define DDIM      256
#define N_EDGES   320000
#define MIN_NORM  1e-15f
#define MAXNORM   0.996f            /* 1 - BALL_EPS, c = 1 */
#define ART_CLIP  (1.0f - 1e-7f)

__device__ __forceinline__ float wave_sum(float v) {
#pragma unroll
  for (int m = 16; m >= 1; m >>= 1) v += __shfl_xor(v, m, 32);
  return v;
}

__device__ __forceinline__ float artanh_c(float x) {
  x = fminf(fmaxf(x, -ART_CLIP), ART_CLIP);
  return 0.5f * (log1pf(x) - log1pf(-x));
}

__device__ __forceinline__ v16h cat8(v8h a, v8h b) {
  return __builtin_shufflevector(a, b, 0, 1, 2, 3, 4, 5, 6, 7,
                                       8, 9, 10, 11, 12, 13, 14, 15);
}

// ---------------------------------------------------------------- K0: bias
__global__ void hgcn_bias(const float* __restrict__ bias, float* __restrict__ hb) {
  const int lane = threadIdx.x;      // 32 threads, one wave
  float b[8]; float s = 0.f;
#pragma unroll
  for (int j = 0; j < 8; ++j) { b[j] = bias[j * 32 + lane]; s += b[j] * b[j]; }
  float bn = fmaxf(sqrtf(wave_sum(s)), MIN_NORM);
  float th = tanhf(bn);
  float sc = th / bn;                               // expmap0
  float n  = fmaxf(th, MIN_NORM);
  float cl = (n > MAXNORM) ? (MAXNORM / n) : 1.f;   // proj
  s = 0.f;
#pragma unroll
  for (int j = 0; j < 8; ++j) {
    float v = sc * cl * b[j];
    hb[j * 32 + lane] = v;
    s += v * v;
  }
  float h2 = wave_sum(s);
  if (lane == 0) hb[DDIM] = h2;                     // ||hyp_bias||^2
}

// ------------------------------------------------ Ks: fp32 -> (hi,lo) fp16
// a = hi + lo with hi = f16(a), lo = f16(a - hi): ~22-bit effective mantissa.
__global__ __launch_bounds__(256)
void hgcn_split(const float* __restrict__ src, int n,
                _Float16* __restrict__ dh, _Float16* __restrict__ dl) {
  int i = blockIdx.x * blockDim.x + threadIdx.x;
  const int stride = gridDim.x * blockDim.x;
  for (; i < n; i += stride) {
    float f = src[i];
    _Float16 h = (_Float16)f;
    dh[i] = h;
    dl[i] = (_Float16)(f - (float)h);
  }
}

// --------------------------------------------------------- K1: WMMA GEMM
// mx[r,o] = sum_k x[r,k] * W[o,k].  One wave per 16x16 output tile.
// Per-lane fragment runs are contiguous in f16 source: b128 loads only.
__global__ __launch_bounds__(256)
void hgcn_gemm_wmma(const _Float16* __restrict__ xh, const _Float16* __restrict__ xl,
                    const _Float16* __restrict__ wh, const _Float16* __restrict__ wl,
                    float* __restrict__ mx) {
  const int lane = threadIdx.x & 31;
  const int wave = threadIdx.x >> 5;
  const int tile = blockIdx.x * 8 + wave;   // 1250 blocks * 8 waves = 10000 tiles
  const int rt   = tile >> 4;               // 625 row tiles
  const int ct   = tile & 15;               // 16  col tiles
  const int subl = lane >> 4;               // lane half (0/1)
  const int ml   = lane & 15;

  // A 16x32 f16 fragment: ah[i] <-> K = kc + 8*subl + i (i<8), kc+16+8*subl+(i-8)
  // B 32x16 f16 fragment: bh[i] <-> K = kc + 8*subl + i (i<8), kc+16+8*subl+(i-8)
  const _Float16* xrh = xh + (size_t)(rt * 16 + ml) * DDIM;
  const _Float16* xrl = xl + (size_t)(rt * 16 + ml) * DDIM;
  const _Float16* wrh = wh + (size_t)(ct * 16 + ml) * DDIM;
  const _Float16* wrl = wl + (size_t)(ct * 16 + ml) * DDIM;
  const int aoff = 8 * subl;
  const int boff = 8 * subl;

  v8f acc = {};
#pragma unroll
  for (int kc = 0; kc < DDIM; kc += 32) {
    v16h Ah = cat8(*(const v8h*)(xrh + kc + aoff),
                   *(const v8h*)(xrh + kc + 16 + aoff));
    v16h Al = cat8(*(const v8h*)(xrl + kc + aoff),
                   *(const v8h*)(xrl + kc + 16 + aoff));
    v16h Bh = cat8(*(const v8h*)(wrh + kc + boff),
                   *(const v8h*)(wrh + kc + 16 + boff));
    v16h Bl = cat8(*(const v8h*)(wrl + kc + boff),
                   *(const v8h*)(wrl + kc + 16 + boff));
    // (Ah+Al)*(Bh+Bl) ~= Ah*Bh + Ah*Bl + Al*Bh   (drop lo*lo, rel err ~2^-22)
    acc = __builtin_amdgcn_wmma_f32_16x16x32_f16(false, Ah, false, Bh, (short)0, acc, false, false);
    acc = __builtin_amdgcn_wmma_f32_16x16x32_f16(false, Ah, false, Bl, (short)0, acc, false, false);
    acc = __builtin_amdgcn_wmma_f32_16x16x32_f16(false, Al, false, Bh, (short)0, acc, false, false);
  }
  // C/D layout: VGPR v -> row M = v + 8*subl, col N = lane%16
#pragma unroll
  for (int v = 0; v < 8; ++v) {
    int row = rt * 16 + v + (subl << 3);
    mx[(size_t)row * DDIM + ct * 16 + ml] = acc[v];
  }
}

// ------------------------------------------- K2: mobius matvec/add + logmap0
// One wave per node row; 8 dims per lane (stride-32 coalesced).
__global__ __launch_bounds__(256)
void hgcn_hyplinear_post(const float* __restrict__ x, const float* __restrict__ hb,
                         float* __restrict__ mxt /* in: mx, out: x_t */) {
  const int lane = threadIdx.x & 31;
  const int row  = blockIdx.x * 8 + (threadIdx.x >> 5);
  const size_t base = (size_t)row * DDIM;

  float xv[8], mv[8], hbv[8];
  float sx = 0.f, sm = 0.f;
#pragma unroll
  for (int j = 0; j < 8; ++j) {
    int d = j * 32 + lane;
    xv[j]  = x[base + d];
    mv[j]  = mxt[base + d];
    hbv[j] = hb[d];
    sx += xv[j] * xv[j];
    sm += mv[j] * mv[j];
  }
  float xn2  = wave_sum(sx);
  float mxn2 = wave_sum(sm);
  float xn   = fmaxf(sqrtf(xn2),  MIN_NORM);
  float mxn  = fmaxf(sqrtf(mxn2), MIN_NORM);

  // mobius_matvec result: tanh(mxn/xn * artanh(xn)) * mx / mxn  (c = 1)
  float t  = tanhf(mxn / xn * artanh_c(xn));
  float sc = (mxn2 == 0.f) ? 0.f : (t / mxn);

  float r[8]; float s = 0.f;
#pragma unroll
  for (int j = 0; j < 8; ++j) { r[j] = sc * mv[j]; s += r[j] * r[j]; }
  float rn = fmaxf(sqrtf(wave_sum(s)), MIN_NORM);
  float cl = (rn > MAXNORM) ? (MAXNORM / rn) : 1.f;   // proj

  s = 0.f; float sxy = 0.f;
#pragma unroll
  for (int j = 0; j < 8; ++j) {
    r[j] *= cl;
    s   += r[j] * r[j];
    sxy += r[j] * hbv[j];
  }
  float x2 = wave_sum(s);
  float xy = wave_sum(sxy);
  float y2 = hb[DDIM];                                 // ||hyp_bias||^2

  // mobius_add(r, hb)
  float ca  = 1.f + 2.f * xy + y2;
  float cb  = 1.f - x2;
  float den = fmaxf(1.f + 2.f * xy + x2 * y2, MIN_NORM);

  float h[8]; s = 0.f;
#pragma unroll
  for (int j = 0; j < 8; ++j) {
    h[j] = (ca * r[j] + cb * hbv[j]) / den;
    s += h[j] * h[j];
  }
  float hn  = fmaxf(sqrtf(wave_sum(s)), MIN_NORM);
  float cl2 = (hn > MAXNORM) ? (MAXNORM / hn) : 1.f;   // proj
  float hnc = fminf(hn, MAXNORM);
  float lg  = artanh_c(hnc) / hnc;                     // logmap0 scale
#pragma unroll
  for (int j = 0; j < 8; ++j)
    mxt[base + j * 32 + lane] = lg * (h[j] * cl2);
}

// ------------------------------------------------- K3: deterministic segment sum
// adj_row is sorted: one block per node finds its edge range via binary search.
__global__ __launch_bounds__(256)
void hgcn_agg(const float* __restrict__ xt, const float* __restrict__ val,
              const int* __restrict__ rows, const int* __restrict__ cols,
              float* __restrict__ out) {
  const int row = blockIdx.x;
  const int d   = threadIdx.x;
  int lo = 0, hi = N_EDGES;
  while (lo < hi) { int mid = (lo + hi) >> 1; if (rows[mid] <  row) lo = mid + 1; else hi = mid; }
  const int start = lo;
  hi = N_EDGES;
  while (lo < hi) { int mid = (lo + hi) >> 1; if (rows[mid] <= row) lo = mid + 1; else hi = mid; }
  const int end = lo;
  float acc = 0.f;
  for (int e = start; e < end; ++e)
    acc += val[e] * xt[(size_t)cols[e] * DDIM + d];    // x_t L2-resident (10 MB)
  out[(size_t)row * DDIM + d] = acc;
}

// ------------------------------------------------- K4: HypAgg expmap + HypAct
__global__ __launch_bounds__(256)
void hgcn_act(float* __restrict__ out) {
  const int lane = threadIdx.x & 31;
  const int row  = blockIdx.x * 8 + (threadIdx.x >> 5);
  const size_t base = (size_t)row * DDIM;

  float u[8]; float s = 0.f;
#pragma unroll
  for (int j = 0; j < 8; ++j) { u[j] = out[base + j * 32 + lane]; s += u[j] * u[j]; }
  float un = fmaxf(sqrtf(wave_sum(s)), MIN_NORM);
  float th = tanhf(un);
  float sc = th / un;                                   // expmap0
  float hn = fmaxf(th, MIN_NORM);                       // ||h|| = tanh(un)
  float cl = (hn > MAXNORM) ? (MAXNORM / hn) : 1.f;     // proj
  float hnc = fminf(hn, MAXNORM);
  float lg  = artanh_c(hnc) / hnc;                      // logmap0

  float xv[8]; s = 0.f;
#pragma unroll
  for (int j = 0; j < 8; ++j) {
    float xt = fmaxf(lg * (sc * cl * u[j]), 0.f);       // relu in tangent space
    xv[j] = xt;
    s += xt * xt;
  }
  float xtn = fmaxf(sqrtf(wave_sum(s)), MIN_NORM);
  float th2 = tanhf(xtn);
  float sc2 = th2 / xtn;                                // expmap0 (c_out = 1)
  float on  = fmaxf(th2, MIN_NORM);
  float cl2 = (on > MAXNORM) ? (MAXNORM / on) : 1.f;    // proj
#pragma unroll
  for (int j = 0; j < 8; ++j)
    out[base + j * 32 + lane] = sc2 * cl2 * xv[j];
}

// ----------------------------------------------------------------- launch
extern "C" void kernel_launch(void* const* d_in, const int* in_sizes, int n_in,
                              void* d_out, int out_size, void* d_ws, size_t ws_size,
                              hipStream_t stream) {
  (void)in_sizes; (void)n_in; (void)out_size; (void)ws_size;
  const float* x       = (const float*)d_in[0];   // [10000,256]
  const float* w       = (const float*)d_in[1];   // [256,256]
  const float* bias    = (const float*)d_in[2];   // [256]
  const float* adj_val = (const float*)d_in[3];   // [320000]
  const int*   adj_row = (const int*)d_in[4];     // [320000] sorted
  const int*   adj_col = (const int*)d_in[5];     // [320000]
  float* out = (float*)d_out;                     // [10000,256]

  // ws layout:
  //   hb  : 512 floats (hyp_bias[256], ||hb||^2 at [256])
  //   xt  : N_NODES*DDIM floats  (mx, then x_t in-place)       ~10.24 MB
  //   xh/xl : N_NODES*DDIM f16 each                            ~5.12 MB each
  //   wh/wl : DDIM*DDIM  f16 each                              128 KB each
  float*    hb = (float*)d_ws;
  float*    xt = hb + 512;
  _Float16* xh = (_Float16*)(xt + (size_t)N_NODES * DDIM);
  _Float16* xl = xh + (size_t)N_NODES * DDIM;
  _Float16* wh = xl + (size_t)N_NODES * DDIM;
  _Float16* wl = wh + (size_t)DDIM * DDIM;

  hgcn_bias<<<1, 32, 0, stream>>>(bias, hb);
  hgcn_split<<<2500, 256, 0, stream>>>(x, N_NODES * DDIM, xh, xl);
  hgcn_split<<<64, 256, 0, stream>>>(w, DDIM * DDIM, wh, wl);
  hgcn_gemm_wmma<<<1250, 256, 0, stream>>>(xh, xl, wh, wl, xt);  // 10000 tiles
  hgcn_hyplinear_post<<<1250, 256, 0, stream>>>(x, hb, xt);      // 10000 rows
  hgcn_agg<<<N_NODES, 256, 0, stream>>>(xt, adj_val, adj_row, adj_col, out);
  hgcn_act<<<1250, 256, 0, stream>>>(out);                       // 10000 rows
}
